// CausalSelfAttention_74569222193319
// MI455X (gfx1250) — matrix operand order repaired and checked by reading, not compiled, not run on hardware
//
#include <hip/hip_runtime.h>
#include <hip/hip_bf16.h>

// ---------------------------------------------------------------------------
// MLA causal self-attention for MI455X (gfx1250), bf16 WMMA pipeline.
// ---------------------------------------------------------------------------

#define T_SEQ   2048
#define N_EMBD  2048
#define N_HEAD  16
#define HEAD_SZ 128
#define KV_LR   512
#define Q_LR    1024
#define ROPE_HS 64
#define QK_HS   (KV_LR + ROPE_HS)          // 576
#define ENC_N   (KV_LR + ROPE_HS + Q_LR)   // 1600
#define QDEC_N  (N_HEAD * QK_HS)           // 9216

typedef __attribute__((ext_vector_type(16))) __bf16 v16bf;
typedef __attribute__((ext_vector_type(8)))  float  v8f;

union FragAB {
    v16bf v;
    uint4 u[2];
    unsigned short s[16];
};

__device__ inline unsigned short f32_to_bf16(float f) {
    union { float f; unsigned u; } x; x.f = f;
    unsigned r = x.u + 0x7FFFu + ((x.u >> 16) & 1u);   // round-to-nearest-even
    return (unsigned short)(r >> 16);
}
__device__ inline float bf16_to_f32(unsigned short s) {
    union { unsigned u; float f; } x; x.u = ((unsigned)s) << 16; return x.f;
}

__device__ inline v8f wmma_bf16(v16bf a, v16bf b, v8f c) {
    // D(16x16,f32) = A(16x32,bf16) x B(32x16,bf16) + C
    return __builtin_amdgcn_wmma_f32_16x16x32_bf16(
        false, a, false, b, (short)0, c, false, false);
}

// A fragment: 16x32 bf16, row-major source, ld in elements.
// lane<16 : row = r0+lane, K = k0+0..7 (v0-3), k0+16..23 (v4-7)
// lane>=16: row = r0+lane-16, K = k0+8..15, k0+24..31
__device__ inline void load_a(FragAB& f, const unsigned short* A, int ld, int r0, int k0) {
    int lane = threadIdx.x & 31;
    const unsigned short* p = A + (size_t)(r0 + (lane & 15)) * ld + k0 + ((lane >> 4) << 3);
    f.u[0] = *(const uint4*)p;
    f.u[1] = *(const uint4*)(p + 16);
}

// B fragment: 32x16 bf16, B stored transposed as BT[N][K], ld = K stride.
// lane<16 : col = c0+lane, K = k0+0..7 (v0-3), k0+16..23 (v4-7)
// lane>=16: col = c0+lane-16, K = k0+8..15, k0+24..31
__device__ inline void load_b(FragAB& f, const unsigned short* BT, int ld, int c0, int k0) {
    int lane = threadIdx.x & 31;
    const unsigned short* p = BT + (size_t)(c0 + (lane & 15)) * ld + k0 + ((lane >> 4) << 3);
    f.u[0] = *(const uint4*)p;
    f.u[1] = *(const uint4*)(p + 16);
}

// C/D fragment: lane<16 -> col=c0+lane, rows r0+i; lane>=16 -> rows r0+8+i.
template <bool OUT_BF16>
__device__ inline void store_c(void* Cv, long cOff, int ldc, int r0, int c0, v8f acc) {
    int lane = threadIdx.x & 31;
    int cc = c0 + (lane & 15);
    int rr = r0 + ((lane >> 4) << 3);
    if (OUT_BF16) {
        unsigned short* C = (unsigned short*)Cv + cOff;
#pragma unroll
        for (int i = 0; i < 8; ++i) C[(size_t)(rr + i) * ldc + cc] = f32_to_bf16(acc[i]);
    } else {
        float* C = (float*)Cv + cOff;
#pragma unroll
        for (int i = 0; i < 8; ++i) C[(size_t)(rr + i) * ldc + cc] = acc[i];
    }
}

// ---------------------------------------------------------------------------
// Conversion kernels
// ---------------------------------------------------------------------------
__global__ void k_f32_to_bf16(const float* __restrict__ src,
                              unsigned short* __restrict__ dst, int n) {
    int i = blockIdx.x * 256 + threadIdx.x;
    if (i < n) dst[i] = f32_to_bf16(src[i]);
}

// src: [batch][R][C] f32 row-major -> dst: [batch][C][R] bf16
__global__ __launch_bounds__(256) void k_transpose_bf16(const float* __restrict__ src,
                                                        unsigned short* __restrict__ dst,
                                                        int R, int C) {
    __shared__ float tile[32][33];
    size_t bs = (size_t)blockIdx.z * R * C;
    src += bs; dst += bs;
    int r0 = blockIdx.y * 32, c0 = blockIdx.x * 32;
    int tx = threadIdx.x & 31, ty = threadIdx.x >> 5;   // 8 row-groups
#pragma unroll
    for (int i = ty; i < 32; i += 8) {
        int r = r0 + i, c = c0 + tx;
        tile[i][tx] = (r < R && c < C) ? src[(size_t)r * C + c] : 0.f;
    }
    __syncthreads();
#pragma unroll
    for (int i = ty; i < 32; i += 8) {
        int c = c0 + i, r = r0 + tx;
        if (c < C && r < R) dst[(size_t)c * R + r] = f32_to_bf16(tile[tx][i]);
    }
}

// ---------------------------------------------------------------------------
// Generic bf16 GEMM: C[M,N] = A[M,K] * B[K,N], with B pre-transposed (BT[N][K]).
// Block = 128 threads = 4 waves (M-stacked). Wave tile 32x64 (2x4 frags):
// 8 v_wmma per K=32 step against 6 fragment loads. Block tile 128x64.
// M % 128 == 0, N % 64 == 0, K % 32 == 0 (true for all call sites).
// ---------------------------------------------------------------------------
template <bool OUT_BF16>
__global__ __launch_bounds__(128) void k_gemm(const unsigned short* __restrict__ A,
                                              const unsigned short* __restrict__ BT,
                                              void* __restrict__ Cv,
                                              int M, int N, int K, int ldc,
                                              long aBS, long bBS, long cBS) {
    (void)M; (void)N;
    int b = blockIdx.z;
    A  += (size_t)b * aBS;
    BT += (size_t)b * bBS;
    long cOff = (long)b * cBS;

    int wid = threadIdx.x >> 5;                 // 0..3 -> stacked along M
    int m0 = blockIdx.y * 128 + wid * 32;
    int n0 = blockIdx.x * 64;

    v8f acc[2][4];
#pragma unroll
    for (int i = 0; i < 2; ++i)
#pragma unroll
        for (int j = 0; j < 4; ++j) acc[i][j] = {};

    for (int k0 = 0; k0 < K; k0 += 32) {
        FragAB a0, a1, bf[4];
        load_a(a0, A, K, m0, k0);
        load_a(a1, A, K, m0 + 16, k0);
#pragma unroll
        for (int j = 0; j < 4; ++j) load_b(bf[j], BT, K, n0 + j * 16, k0);
#pragma unroll
        for (int j = 0; j < 4; ++j) {
            acc[0][j] = wmma_bf16(a0.v, bf[j].v, acc[0][j]);
            acc[1][j] = wmma_bf16(a1.v, bf[j].v, acc[1][j]);
        }
    }
#pragma unroll
    for (int i = 0; i < 2; ++i)
#pragma unroll
        for (int j = 0; j < 4; ++j)
            store_c<OUT_BF16>(Cv, cOff, ldc, m0 + i * 16, n0 + j * 16, acc[i][j]);
}

// ---------------------------------------------------------------------------
// prep: enc f32 [T,1600] -> keq bf16 [T,576] (RoPE on cols 512..575),
//       vT bf16 [512][T] (transposed c_kv), cq bf16 [T,1024]
// ---------------------------------------------------------------------------
__global__ void k_prep(const float* __restrict__ enc,
                       const float* __restrict__ cosb, const float* __restrict__ sinb,
                       unsigned short* __restrict__ keq,
                       unsigned short* __restrict__ vT,
                       unsigned short* __restrict__ cq) {
    int t = blockIdx.x;
    const float* e = enc + (size_t)t * ENC_N;
    for (int j = threadIdx.x; j < ENC_N; j += blockDim.x) {
        if (j < KV_LR) {
            unsigned short bv = f32_to_bf16(e[j]);
            keq[(size_t)t * QK_HS + j] = bv;
            vT[(size_t)j * T_SEQ + t]  = bv;
        } else if (j < QK_HS) {
            int jj = j - KV_LR;
            float c = cosb[t * ROPE_HS + jj], s = sinb[t * ROPE_HS + jj];
            float other = (jj < 32) ? -e[KV_LR + jj + 32] : e[KV_LR + jj - 32];
            keq[(size_t)t * QK_HS + j] = f32_to_bf16(e[j] * c + other * s);
        } else {
            cq[(size_t)t * Q_LR + (j - QK_HS)] = f32_to_bf16(e[j]);
        }
    }
}

// ---------------------------------------------------------------------------
// rope_q: qtmp bf16 [T, 16*576] -> qh bf16 [16][T][576], RoPE on last 64 dims,
// pre-scaled by 1/sqrt(HEAD_SZ) so logits come out of WMMA already scaled.
// ---------------------------------------------------------------------------
__global__ void k_rope_q(const unsigned short* __restrict__ qtmp,
                         const float* __restrict__ cosb, const float* __restrict__ sinb,
                         unsigned short* __restrict__ qh) {
    int t = blockIdx.x, h = blockIdx.y;
    const unsigned short* src = qtmp + (size_t)t * QDEC_N + (size_t)h * QK_HS;
    unsigned short* dst = qh + ((size_t)h * T_SEQ + t) * QK_HS;
    const float scale = 0.08838834764831845f;   // 1/sqrt(128)
    for (int j = threadIdx.x; j < QK_HS; j += blockDim.x) {
        float v;
        if (j < KV_LR) {
            v = bf16_to_f32(src[j]);
        } else {
            int jj = j - KV_LR;
            float c = cosb[t * ROPE_HS + jj], s = sinb[t * ROPE_HS + jj];
            float other = (jj < 32) ? -bf16_to_f32(src[KV_LR + jj + 32])
                                    :  bf16_to_f32(src[KV_LR + jj - 32]);
            v = bf16_to_f32(src[j]) * c + other * s;
        }
        dst[j] = f32_to_bf16(v * scale);
    }
}

// ---------------------------------------------------------------------------
// Flash attention: grid (T/16, N_HEAD), block = 64 (2 waves).
// Each wave: 16 query rows x 256 V-dims (wave picks half of the 512 dims),
// online softmax over 32-key blocks. 16 f32 acc frags resident in VGPRs.
// ---------------------------------------------------------------------------
__device__ inline float half_red_max(float v) {
#pragma unroll
    for (int o = 1; o < 16; o <<= 1) v = fmaxf(v, __shfl_xor(v, o, 16));
    return v;
}
__device__ inline float half_red_sum(float v) {
#pragma unroll
    for (int o = 1; o < 16; o <<= 1) v += __shfl_xor(v, o, 16);
    return v;
}

__global__ __launch_bounds__(64) void k_attn(const unsigned short* __restrict__ qh,
                                             const unsigned short* __restrict__ keq,
                                             const unsigned short* __restrict__ vT,
                                             unsigned short* __restrict__ yh) {
    alignas(16) __shared__ unsigned short pbuf[2][16][40];  // per-wave P tile (16x32 bf16)

    int qt = blockIdx.x, h = blockIdx.y;
    int wave = threadIdx.x >> 5;
    int lane = threadIdx.x & 31;
    int lhalf = lane >> 4;        // 0: rows 0..7 / cols K0..; 1: rows 8..15
    int l15 = lane & 15;

    const unsigned short* Q = qh + (size_t)h * T_SEQ * QK_HS;
    int qrow0 = qt * 16;

    v8f acc[16];
#pragma unroll
    for (int f = 0; f < 16; ++f) acc[f] = {};
    float rowmax[8], rowsum[8];
#pragma unroll
    for (int i = 0; i < 8; ++i) { rowmax[i] = -3.0e38f; rowsum[i] = 0.f; }

    int nkb = ((qrow0 + 15) >> 5) + 1;      // 32-key blocks needed for causal range
    for (int kb = 0; kb < nkb; ++kb) {
        int kc0 = kb * 32;

        // ---- S = Q K^T (scaled): two 16x16 f32 frags over 32 keys ----
        v8f s0 = {}, s1 = {};
        for (int ks = 0; ks < QK_HS / 32; ++ks) {
            int k0 = ks * 32;
            FragAB a, b0, b1;
            load_a(a, Q, QK_HS, qrow0, k0);
            load_b(b0, keq, QK_HS, kc0, k0);
            load_b(b1, keq, QK_HS, kc0 + 16, k0);
            s0 = wmma_bf16(a.v, b0.v, s0);
            s1 = wmma_bf16(a.v, b1.v, s1);
        }

        // ---- causal mask + online softmax (per-row stats in half-waves) ----
        float alpha[8];
#pragma unroll
        for (int i = 0; i < 8; ++i) {
            int row = qrow0 + i + lhalf * 8;
            float v0 = (kc0 + l15 <= row)      ? (float)s0[i] : -3.0e38f;
            float v1 = (kc0 + 16 + l15 <= row) ? (float)s1[i] : -3.0e38f;
            float bm = half_red_max(fmaxf(v0, v1));
            float nm = fmaxf(rowmax[i], bm);
            alpha[i] = __expf(rowmax[i] - nm);
            rowmax[i] = nm;
            float p0 = __expf(v0 - nm);
            float p1 = __expf(v1 - nm);
            s0[i] = p0; s1[i] = p1;
            rowsum[i] = rowsum[i] * alpha[i] + half_red_sum(p0 + p1);
        }
#pragma unroll
        for (int f = 0; f < 16; ++f)
#pragma unroll
            for (int i = 0; i < 8; ++i) acc[f][i] *= alpha[i];

        // ---- re-layout P (C-frag) -> A-frag via per-wave LDS tile ----
        unsigned short (*pb)[40] = pbuf[wave];
#pragma unroll
        for (int i = 0; i < 8; ++i) {
            int r = i + lhalf * 8;
            pb[r][l15]      = f32_to_bf16(s0[i]);
            pb[r][l15 + 16] = f32_to_bf16(s1[i]);
        }
        FragAB pf;
        {
            const unsigned short* pr = &pb[l15][0] + (lhalf << 3);
            pf.u[0] = *(const uint4*)pr;
            pf.u[1] = *(const uint4*)(pr + 16);
        }

        // ---- acc += P * V over this 32-key block (B frags from vT[dim][t]) ----
#pragma unroll
        for (int f = 0; f < 16; ++f) {
            int d0 = wave * 256 + f * 16;
            FragAB bv;
            load_b(bv, vT, T_SEQ, d0, kc0);
            acc[f] = wmma_bf16(pf.v, bv.v, acc[f]);
        }
    }

    // ---- normalize and store yh[h][t][512] as bf16 ----
    float inv[8];
#pragma unroll
    for (int i = 0; i < 8; ++i) inv[i] = 1.f / rowsum[i];
    unsigned short* Y = yh + (size_t)h * T_SEQ * KV_LR;
#pragma unroll
    for (int f = 0; f < 16; ++f) {
        int d0 = wave * 256 + f * 16 + l15;
#pragma unroll
        for (int i = 0; i < 8; ++i) {
            int r = qrow0 + i + lhalf * 8;
            Y[(size_t)r * KV_LR + d0] = f32_to_bf16(acc[f][i] * inv[i]);
        }
    }
}

// ---------------------------------------------------------------------------
// Host launcher
// ---------------------------------------------------------------------------
extern "C" void kernel_launch(void* const* d_in, const int* in_sizes, int n_in,
                              void* d_out, int out_size, void* d_ws, size_t ws_size,
                              hipStream_t stream) {
    (void)in_sizes; (void)n_in; (void)out_size; (void)ws_size;
    const float* x     = (const float*)d_in[0];
    const float* cosb  = (const float*)d_in[1];
    const float* sinb  = (const float*)d_in[2];
    const float* Wqkv  = (const float*)d_in[3];
    const float* Wqdec = (const float*)d_in[4];
    const float* projv = (const float*)d_in[5];
    const float* Wout  = (const float*)d_in[6];
    float* out = (float*)d_out;

    char* ws = (char*)d_ws;
    size_t off = 0;
    auto alloc = [&](size_t bytes) { size_t o = off; off += (bytes + 255) & ~(size_t)255; return o; };

    const size_t oXB     = alloc((size_t)T_SEQ * N_EMBD * 2);        // xb; reused later as yv
    const size_t oWQKVT  = alloc((size_t)ENC_N * N_EMBD * 2);
    const size_t oWQDECT = alloc((size_t)QDEC_N * Q_LR * 2);
    const size_t oPROJVT = alloc((size_t)N_HEAD * HEAD_SZ * KV_LR * 2);
    const size_t oWOUTT  = alloc((size_t)N_EMBD * N_EMBD * 2);
    const size_t oENC    = alloc((size_t)T_SEQ * ENC_N * 4);
    const size_t oKEQ    = alloc((size_t)T_SEQ * QK_HS * 2);
    const size_t oVT     = alloc((size_t)KV_LR * T_SEQ * 2);
    const size_t oCQ     = alloc((size_t)T_SEQ * Q_LR * 2);
    const size_t oQTMP   = alloc((size_t)T_SEQ * QDEC_N * 2);        // reused later as yh
    const size_t oQH     = alloc((size_t)N_HEAD * T_SEQ * QK_HS * 2);

    unsigned short* XB     = (unsigned short*)(ws + oXB);
    unsigned short* WQKVT  = (unsigned short*)(ws + oWQKVT);
    unsigned short* WQDECT = (unsigned short*)(ws + oWQDECT);
    unsigned short* PROJVT = (unsigned short*)(ws + oPROJVT);
    unsigned short* WOUTT  = (unsigned short*)(ws + oWOUTT);
    float*          ENC    = (float*)(ws + oENC);
    unsigned short* KEQ    = (unsigned short*)(ws + oKEQ);
    unsigned short* VT     = (unsigned short*)(ws + oVT);
    unsigned short* CQ     = (unsigned short*)(ws + oCQ);
    unsigned short* QTMP   = (unsigned short*)(ws + oQTMP);
    unsigned short* QH     = (unsigned short*)(ws + oQH);
    unsigned short* YH     = QTMP;   // qtmp dead after rope_q
    unsigned short* YV     = XB;     // xb dead after GEMM1

    // --- input conversions ---
    {
        int n = T_SEQ * N_EMBD;
        k_f32_to_bf16<<<(n + 255) / 256, 256, 0, stream>>>(x, XB, n);
    }
    k_transpose_bf16<<<dim3(ENC_N / 32, N_EMBD / 32, 1), 256, 0, stream>>>(Wqkv, WQKVT, N_EMBD, ENC_N);
    k_transpose_bf16<<<dim3(QDEC_N / 32, Q_LR / 32, 1), 256, 0, stream>>>(Wqdec, WQDECT, Q_LR, QDEC_N);
    k_transpose_bf16<<<dim3(HEAD_SZ / 32, KV_LR / 32, N_HEAD), 256, 0, stream>>>(projv, PROJVT, KV_LR, HEAD_SZ);
    k_transpose_bf16<<<dim3(N_EMBD / 32, N_EMBD / 32, 1), 256, 0, stream>>>(Wout, WOUTT, N_EMBD, N_EMBD);

    // --- GEMM1: enc = x @ W_qkv  (f32 out) ---
    k_gemm<false><<<dim3(ENC_N / 64, T_SEQ / 128, 1), 128, 0, stream>>>(
        XB, WQKVT, ENC, T_SEQ, ENC_N, N_EMBD, ENC_N, 0, 0, 0);

    // --- split + rope(k_r) + transpose(v) ---
    k_prep<<<T_SEQ, 256, 0, stream>>>(ENC, cosb, sinb, KEQ, VT, CQ);

    // --- GEMM2: q = c_q @ W_qdec (bf16 out) ---
    k_gemm<true><<<dim3(QDEC_N / 64, T_SEQ / 128, 1), 128, 0, stream>>>(
        CQ, WQDECT, QTMP, T_SEQ, QDEC_N, Q_LR, QDEC_N, 0, 0, 0);

    // --- rope(q_r) + per-head layout + 1/sqrt(d) pre-scale ---
    k_rope_q<<<dim3(T_SEQ, N_HEAD), 256, 0, stream>>>(QTMP, cosb, sinb, QH);

    // --- flash attention ---
    k_attn<<<dim3(T_SEQ / 16, N_HEAD), 64, 0, stream>>>(QH, KEQ, VT, YH);

    // --- GEMM3 (batched per head): yv[:, h*128:(h+1)*128] = yh[h] @ proj_v[h] (bf16 out) ---
    k_gemm<true><<<dim3(HEAD_SZ / 64, T_SEQ / 128, N_HEAD), 128, 0, stream>>>(
        YH, PROJVT, YV, T_SEQ, HEAD_SZ, KV_LR, N_EMBD,
        (long)T_SEQ * KV_LR, (long)HEAD_SZ * KV_LR, (long)HEAD_SZ);

    // --- GEMM4: out = yv @ W_out (f32 out) ---
    k_gemm<false><<<dim3(N_EMBD / 64, T_SEQ / 128, 1), 128, 0, stream>>>(
        YV, WOUTT, out, T_SEQ, N_EMBD, N_EMBD, N_EMBD, 0, 0, 0);
}
